// GCGNet_27702539059792
// MI455X (gfx1250) — matrix operand order repaired and checked by reading, not compiled, not run on hardware
//
#include <hip/hip_runtime.h>
#include <hip/hip_bf16.h>
#include <math.h>

// ---------------- problem constants ----------------
#define NN 25600      // nodes
#define NE 409600     // edges
#define NB 512        // graphs
#define NPG 50        // nodes per graph (contiguous)
#define FXD 78
#define FXT 954

typedef __bf16 bf16_t;
typedef __attribute__((ext_vector_type(16))) __bf16 v16bf;
typedef __attribute__((ext_vector_type(8)))  float  v8f;

// ---------------- elementwise / utility kernels ----------------

__global__ void k_fzero(float* p, long long n) {
    long long i = (long long)blockIdx.x * blockDim.x + threadIdx.x;
    if (i < n) p[i] = 0.0f;
}

// Pack fp32 A[M,K] -> bf16 WMMA A-fragments, layout [M/16][padK/32][lane(32)][elt(16)]
// ISA 7.12.2 (16-bit A 16x32): lane = khalf*16 + m ;
//   frag pair v<4 : K = khalf*8 + 2v ; v>=4 : K = 16 + khalf*8 + 2(v-4)
__global__ void k_pack_A(const float* __restrict__ src, bf16_t* __restrict__ dst,
                         int M, int K, int numKt) {
    long long i = (long long)blockIdx.x * blockDim.x + threadIdx.x;
    long long total = (long long)(M / 16) * numKt * 512;
    if (i >= total) return;
    int e    = (int)(i & 15);
    int lane = (int)((i >> 4) & 31);
    long long t = i >> 9;                       // mt*numKt + kt
    int kt = (int)(t % numKt);
    int mt = (int)(t / numKt);
    int m = mt * 16 + (lane & 15);
    int khalf = lane >> 4;
    int v = e >> 1, odd = e & 1;
    int kk = (v < 4) ? (khalf * 8 + 2 * v) : (16 + khalf * 8 + 2 * (v - 4));
    int k = kt * 32 + kk + odd;
    float val = (k < K) ? src[(long long)m * K + k] : 0.0f;
    dst[i] = (bf16_t)val;
}

// Pack fp32 W[K,N] -> bf16 WMMA B-fragments, layout [padN/16][padK/32][lane(32)][elt(16)]
// B (32x16): same K order as A: v<4 : K = khalf*8 + 2v ; v>=4 : K = 16 + khalf*8 + 2(v-4)
__global__ void k_pack_B(const float* __restrict__ src, bf16_t* __restrict__ dst,
                         int K, int N, int numKt, int numNt) {
    long long i = (long long)blockIdx.x * blockDim.x + threadIdx.x;
    long long total = (long long)numNt * numKt * 512;
    if (i >= total) return;
    int e    = (int)(i & 15);
    int lane = (int)((i >> 4) & 31);
    long long t = i >> 9;                       // nt*numKt + kt
    int kt = (int)(t % numKt);
    int nt = (int)(t / numKt);
    int n = nt * 16 + (lane & 15);
    int khalf = lane >> 4;
    int v = e >> 1, odd = e & 1;
    int k = kt * 32 + ((v < 4) ? (khalf * 8 + 2 * v) : (16 + khalf * 8 + 2 * (v - 4))) + odd;
    float val = (k < K && n < N) ? src[(long long)k * N + n] : 0.0f;
    dst[i] = (bf16_t)val;
}

__global__ void k_deg_count(const int* __restrict__ tgt, float* __restrict__ deg, int nE) {
    int e = blockIdx.x * blockDim.x + threadIdx.x;
    if (e < nE) atomicAdd(&deg[tgt[e]], 1.0f);
}

__global__ void k_make_dinv(const float* __restrict__ deg, float* __restrict__ dinv, int n) {
    int i = blockIdx.x * blockDim.x + threadIdx.x;
    if (i < n) dinv[i] = rsqrtf(deg[i] + 1.0f);
}

// agg[tgt] += h[src] * dinv[src]*dinv[tgt], one thread per (edge, feature)
__global__ void k_gcn_scatter(const float* __restrict__ h, const int* __restrict__ src,
                              const int* __restrict__ tgt, const float* __restrict__ dinv,
                              float* __restrict__ agg, int nE, int F) {
    long long i = (long long)blockIdx.x * blockDim.x + threadIdx.x;
    if (i >= (long long)nE * F) return;
    int e = (int)(i / F);
    int f = (int)(i % F);
    int s = src[e], t = tgt[e];
    float coef = dinv[s] * dinv[t];
    atomicAdd(&agg[(long long)t * F + f], h[(long long)s * F + f] * coef);
}

// x = relu(agg + h * dinv[n]^2 + b[f])   (in place on agg)
__global__ void k_gcn_finalize(float* __restrict__ agg, const float* __restrict__ h,
                               const float* __restrict__ dinv, const float* __restrict__ b,
                               int nNodes, int F) {
    long long i = (long long)blockIdx.x * blockDim.x + threadIdx.x;
    if (i >= (long long)nNodes * F) return;
    int n = (int)(i / F);
    int f = (int)(i % F);
    float d = dinv[n];
    float v = agg[i] + h[i] * d * d + b[f];
    agg[i] = fmaxf(v, 0.0f);
}

// segment max over contiguous 50-node graphs
__global__ void k_pool_max(const float* __restrict__ x, float* __restrict__ o, int F) {
    long long i = (long long)blockIdx.x * blockDim.x + threadIdx.x;
    if (i >= (long long)NB * F) return;
    int g = (int)(i / F);
    int f = (int)(i % F);
    const float* base = x + (long long)g * NPG * F + f;
    float m = base[0];
    for (int k = 1; k < NPG; ++k) m = fmaxf(m, base[(long long)k * F]);
    o[i] = m;
}

// L2 row-normalize: y[r,:] = x[r,:] / max(||x[r,:]||, 1e-12)
__global__ void k_rownorm(const float* __restrict__ x, float* __restrict__ y, int cols) {
    __shared__ float red[256];
    int row = blockIdx.x;
    const float* xr = x + (long long)row * cols;
    float ss = 0.0f;
    for (int c = threadIdx.x; c < cols; c += 256) { float v = xr[c]; ss += v * v; }
    red[threadIdx.x] = ss;
    __syncthreads();
    for (int s = 128; s > 0; s >>= 1) {
        if ((int)threadIdx.x < s) red[threadIdx.x] += red[threadIdx.x + s];
        __syncthreads();
    }
    float inv = 1.0f / fmaxf(sqrtf(red[0]), 1e-12f);
    for (int c = threadIdx.x; c < cols; c += 256) y[(long long)row * cols + c] = xr[c] * inv;
}

__global__ void k_concat2(const float* __restrict__ a, const float* __restrict__ b,
                          float* __restrict__ o, int B, int F) {
    long long i = (long long)blockIdx.x * blockDim.x + threadIdx.x;
    if (i >= (long long)B * 2 * F) return;
    int r = (int)(i / (2 * F));
    int c = (int)(i % (2 * F));
    o[i] = (c < F) ? a[(long long)r * F + c] : b[(long long)r * F + (c - F)];
}

__global__ void k_concat4(const float* __restrict__ a, const float* __restrict__ b,
                          const float* __restrict__ c4, const float* __restrict__ d,
                          float* __restrict__ o, int B, int F) {
    long long i = (long long)blockIdx.x * blockDim.x + threadIdx.x;
    if (i >= (long long)B * 4 * F) return;
    int r = (int)(i / (4 * F));
    int c = (int)(i % (4 * F));
    const float* srcs[4] = {a, b, c4, d};
    o[i] = srcs[c / F][(long long)r * F + (c % F)];
}

// ---------------- WMMA bf16 GEMM on pre-packed fragments ----------------
// Ap: [numMt][numKt][32][16] bf16 (A fragments), Bp: [numNt][numKt][32][16]
// Template NT = N tiles per wave (1 or 2): branch-free inner loop, A fragment
// reused across the NT v_wmma_f32_16x16x32_bf16 per K step. One-deep software
// pipeline: fragments for kt+1 are in flight while wmma executes on kt.
// 4 waves per block; all control flow is blockIdx/wave-uniform -> EXEC all-1s
// at every WMMA. act: 0 = none, 1 = relu, 2 = sigmoid.
template <int NT>
__global__ void k_wmma_gemm(const bf16_t* __restrict__ Ap, const bf16_t* __restrict__ Bp,
                            const float* __restrict__ bias, float* __restrict__ C,
                            int M, int N, int numKt, int ntBase, int act, int hasBias) {
    const int lane = threadIdx.x & 31;
    const int numMt = M / 16;
    const int mt = blockIdx.x * 4 + (threadIdx.x >> 5);
    if (mt >= numMt) return;                       // wave-uniform
    const int nt0 = ntBase + blockIdx.y * NT;

    const v16bf* a  = (const v16bf*)Ap + ((long long)mt  * numKt) * 32 + lane;
    const v16bf* b0 = (const v16bf*)Bp + ((long long)nt0 * numKt) * 32 + lane;
    const v16bf* b1 = b0 + (long long)numKt * 32;

    v8f acc0 = {};
    v8f acc1 = {};

    // prologue of the pipeline
    v16bf af  = a[0];
    v16bf bf0 = b0[0];
    v16bf bf1 = bf0;
    if (NT == 2) bf1 = b1[0];

    for (int kt = 0; kt < numKt; ++kt) {
        // issue next step's fragment loads before this step's WMMAs
        int nk = (kt + 1 < numKt) ? (kt + 1) : kt;   // clamped -> branch-free
        v16bf afn  = a [(long long)nk * 32];
        v16bf bf0n = b0[(long long)nk * 32];
        v16bf bf1n = bf0n;
        if (NT == 2) bf1n = b1[(long long)nk * 32];
        __builtin_prefetch(&a [(long long)(nk + 1) * 32], 0, 3);
        __builtin_prefetch(&b0[(long long)(nk + 1) * 32], 0, 3);

        acc0 = __builtin_amdgcn_wmma_f32_16x16x32_bf16(
            false, af, false, bf0, (short)0, acc0, false, false);
        if (NT == 2)
            acc1 = __builtin_amdgcn_wmma_f32_16x16x32_bf16(
                false, af, false, bf1, (short)0, acc1, false, false);

        af = afn; bf0 = bf0n;
        if (NT == 2) bf1 = bf1n;
    }

    // C/D layout: lanes 0-15: (M=r, N=lane); lanes 16-31: (M=r+8, N=lane-16)
    const int nloc = lane & 15;
    const int mAdd = (lane < 16) ? 0 : 8;
#pragma unroll
    for (int t = 0; t < NT; ++t) {
        int n = (nt0 + t) * 16 + nloc;
        if (n < N) {
            float bv = hasBias ? bias[n] : 0.0f;
            v8f acc = t ? acc1 : acc0;
#pragma unroll
            for (int r = 0; r < 8; ++r) {
                int m = mt * 16 + r + mAdd;
                float v = acc[r] + bv;
                if (act == 1) v = fmaxf(v, 0.0f);
                else if (act == 2) v = 1.0f / (1.0f + expf(-v));
                C[(long long)m * N + n] = v;
            }
        }
    }
}

// ---------------- host-side helpers ----------------

static inline unsigned blks(long long total, int bs) { return (unsigned)((total + bs - 1) / bs); }

// act: 0 none, 1 relu, 2 sigmoid. bias may be nullptr.
static void dense(const float* A, int M, const float* W, const float* bias,
                  int K, int N, bf16_t* Abf, bf16_t* Wbf, float* C, int act,
                  hipStream_t s) {
    const int padK  = (K + 31) & ~31;
    const int padN  = (N + 15) & ~15;
    const int numKt = padK / 32;
    const int numNt = padN / 16;
    const int numMt = M / 16;
    { long long t = (long long)numNt * numKt * 512;
      k_pack_B<<<blks(t, 256), 256, 0, s>>>(W, Wbf, K, N, numKt, numNt); }
    { long long t = (long long)numMt * numKt * 512;
      k_pack_A<<<blks(t, 256), 256, 0, s>>>(A, Abf, M, K, numKt); }
    const unsigned gx = (unsigned)((numMt + 3) / 4);
    const int numPairs = numNt / 2;
    if (numPairs > 0) {
        dim3 grid(gx, (unsigned)numPairs);
        k_wmma_gemm<2><<<grid, 128, 0, s>>>(Abf, Wbf, bias, C, M, N, numKt, 0, act,
                                            bias != nullptr);
    }
    if (numNt & 1) {                 // odd leftover N tile
        dim3 grid(gx, 1);
        k_wmma_gemm<1><<<grid, 128, 0, s>>>(Abf, Wbf, bias, C, M, N, numKt, numNt - 1, act,
                                            bias != nullptr);
    }
}

extern "C" void kernel_launch(void* const* d_in, const int* in_sizes, int n_in,
                              void* d_out, int out_size, void* d_ws, size_t ws_size,
                              hipStream_t stream) {
    (void)in_sizes; (void)n_in; (void)out_size; (void)ws_size;

    const float* x1   = (const float*)d_in[0];
    const int*   ei1  = (const int*)  d_in[1];
    const float* x2   = (const float*)d_in[3];
    const int*   ei2  = (const int*)  d_in[4];
    const float* cell = (const float*)d_in[6];
    const float* P[24];
    for (int i = 0; i < 24; ++i) P[i] = (const float*)d_in[7 + i];
    const float *conv1_W = P[0],  *conv1_b = P[1];
    const float *conv2_W = P[2],  *conv2_b = P[3];
    const float *conv3_W = P[4],  *conv3_b = P[5];
    const float *fcg1_W  = P[6],  *fcg1_b  = P[7];
    const float *fcg2_W  = P[8],  *fcg2_b  = P[9];
    const float *red1_W  = P[10], *red1_b  = P[11];
    const float *red2_W  = P[12], *red2_b  = P[13];
    const float *red3_W  = P[14], *red3_b  = P[15];
    const float *gate_W  = P[16], *gate_b  = P[17];
    const float *fc1_W   = P[18], *fc1_b   = P[19];
    const float *fc2_W   = P[20], *fc2_b   = P[21];
    const float *out_W   = P[22], *out_b   = P[23];

    // ---- carve workspace ----
    char* w = (char*)d_ws;
    auto take = [&](size_t bytes) { char* p = w; w += (bytes + 255) & ~(size_t)255; return p; };
    bf16_t* BFA    = (bf16_t*)take(sizeof(bf16_t) * (size_t)NN * 320);  // packed A (max M*padK)
    bf16_t* BFW    = (bf16_t*)take(sizeof(bf16_t) * 960 * 512);         // packed B (max padK*padN, red1)
    float*  H      = (float*) take(sizeof(float) * (size_t)NN * 312);
    float*  A0     = (float*) take(sizeof(float) * (size_t)NN * 312);
    float*  A1     = (float*) take(sizeof(float) * (size_t)NN * 312);
    float*  DEG    = (float*) take(sizeof(float) * NN);
    float*  DINV   = (float*) take(sizeof(float) * NN);
    float*  POOLED = (float*) take(sizeof(float) * NB * 312);
    float*  G1OUT  = (float*) take(sizeof(float) * NB * 156);
    float*  D1     = (float*) take(sizeof(float) * NB * 128);
    float*  D2     = (float*) take(sizeof(float) * NB * 128);
    float*  CELLN  = (float*) take(sizeof(float) * NB * FXT);
    float*  CV1    = (float*) take(sizeof(float) * NB * 512);
    float*  CV2    = (float*) take(sizeof(float) * NB * 256);
    float*  CVO    = (float*) take(sizeof(float) * NB * 128);
    float*  CAT2   = (float*) take(sizeof(float) * NB * 256);
    float*  GATEO  = (float*) take(sizeof(float) * NB * 128);
    float*  XC     = (float*) take(sizeof(float) * NB * 512);
    float*  F1O    = (float*) take(sizeof(float) * NB * 512);
    float*  F2O    = (float*) take(sizeof(float) * NB * 128);

    // ---- drug encoders (shared weights, two inputs) ----
    for (int drug = 0; drug < 2; ++drug) {
        const float* x  = drug ? x2 : x1;
        const int*   ei = drug ? ei2 : ei1;
        float*     dvec = drug ? D2 : D1;
        const int* srcI = ei;            // edge_index[0]
        const int* tgtI = ei + NE;       // edge_index[1]

        // degree + dinv
        k_fzero<<<blks(NN, 256), 256, 0, stream>>>(DEG, NN);
        k_deg_count<<<blks(NE, 256), 256, 0, stream>>>(tgtI, DEG, NE);
        k_make_dinv<<<blks(NN, 256), 256, 0, stream>>>(DEG, DINV, NN);

        // ---- conv1: 78 -> 78 ----
        dense(x, NN, conv1_W, nullptr, FXD, FXD, BFA, BFW, H, 0, stream);
        k_fzero<<<blks((long long)NN * 78, 256), 256, 0, stream>>>(A0, (long long)NN * 78);
        k_gcn_scatter<<<blks((long long)NE * 78, 256), 256, 0, stream>>>(H, srcI, tgtI, DINV, A0, NE, 78);
        k_gcn_finalize<<<blks((long long)NN * 78, 256), 256, 0, stream>>>(A0, H, DINV, conv1_b, NN, 78);

        // ---- conv2: 78 -> 156 ----
        dense(A0, NN, conv2_W, nullptr, 78, 156, BFA, BFW, H, 0, stream);
        k_fzero<<<blks((long long)NN * 156, 256), 256, 0, stream>>>(A1, (long long)NN * 156);
        k_gcn_scatter<<<blks((long long)NE * 156, 256), 256, 0, stream>>>(H, srcI, tgtI, DINV, A1, NE, 156);
        k_gcn_finalize<<<blks((long long)NN * 156, 256), 256, 0, stream>>>(A1, H, DINV, conv2_b, NN, 156);

        // ---- conv3: 156 -> 312 ----
        dense(A1, NN, conv3_W, nullptr, 156, 312, BFA, BFW, H, 0, stream);
        k_fzero<<<blks((long long)NN * 312, 256), 256, 0, stream>>>(A0, (long long)NN * 312);
        k_gcn_scatter<<<blks((long long)NE * 312, 256), 256, 0, stream>>>(H, srcI, tgtI, DINV, A0, NE, 312);
        k_gcn_finalize<<<blks((long long)NN * 312, 256), 256, 0, stream>>>(A0, H, DINV, conv3_b, NN, 312);

        // ---- pool + graph MLP ----
        k_pool_max<<<blks((long long)NB * 312, 256), 256, 0, stream>>>(A0, POOLED, 312);
        dense(POOLED, NB, fcg1_W, fcg1_b, 312, 156, BFA, BFW, G1OUT, 1, stream);
        dense(G1OUT,  NB, fcg2_W, fcg2_b, 156, 128, BFA, BFW, dvec,  0, stream);
    }

    // ---- cell line pathway ----
    k_rownorm<<<NB, 256, 0, stream>>>(cell, CELLN, FXT);
    dense(CELLN, NB, red1_W, red1_b, FXT, 512, BFA, BFW, CV1, 1, stream);
    dense(CV1,   NB, red2_W, red2_b, 512, 256, BFA, BFW, CV2, 1, stream);
    dense(CV2,   NB, red3_W, red3_b, 256, 128, BFA, BFW, CVO, 0, stream);

    // ---- gate + head ----
    k_concat2<<<blks((long long)NB * 256, 256), 256, 0, stream>>>(D1, D2, CAT2, NB, 128);
    dense(CAT2, NB, gate_W, gate_b, 256, 128, BFA, BFW, GATEO, 2, stream);   // sigmoid
    k_concat4<<<blks((long long)NB * 512, 256), 256, 0, stream>>>(D1, D2, GATEO, CVO, XC, NB, 128);
    dense(XC,  NB, fc1_W, fc1_b, 512, 512, BFA, BFW, F1O, 1, stream);
    dense(F1O, NB, fc2_W, fc2_b, 512, 128, BFA, BFW, F2O, 1, stream);
    dense(F2O, NB, out_W, out_b, 128, 2,   BFA, BFW, (float*)d_out, 0, stream);
}
